// FlowMatchingAction_5119601017537
// MI455X (gfx1250) — matrix operand order repaired and checked by reading, not compiled, not run on hardware
//
#include <hip/hip_runtime.h>
#include <hip/hip_bf16.h>
#include <math.h>

// ---------------------------------------------------------------------------
// FlowMatchingAction head for MI455X (gfx1250), wave32 + WMMA f16 (16x16x32).
//
// ~104 GFLOP total, weights ~0.5MB f16 (L2-resident), activations on-chip.
// 128 blocks x 8 waves (1024 waves): each block owns a 16-row batch tile,
// each wave owns 2 N-tiles (layers 0-2) / 1 N-tile (layer 3).
// ---------------------------------------------------------------------------

typedef __attribute__((ext_vector_type(16))) _Float16 v16h;
typedef __attribute__((ext_vector_type(8)))  float    v8f;

#define AHD    256
#define FLATD  112
#define STEPS  100
#define NWAVES 8

// -------- A-fragment LDS addressing (f16 16x32 A-matrix layout) ------------
// lanes 0-15 hold K in {0..7, 16..23}, lanes 16-31 hold K in {8..15, 24..31};
// per lane, 16 contiguous halfs cover its two K octets in register order.
// Address (in halfs): chunk(k/32)*512 + lane*16 + h
__device__ __forceinline__ int aaddr(int m, int k) {
  return ((k >> 5) << 9)                               // chunk * 512
       + ((m + (((k >> 3) & 1) << 4)) << 4)            // lane  * 16
       + (((k >> 4) & 1) << 3) + (k & 7);              // h
}

__device__ __forceinline__ float gelu_exact(float x) {
  return 0.5f * x * (1.0f + erff(x * 0.70710678118654752440f));
}

// -------- weight pre-pack: fp32 (K,N) row-major -> f16 B-fragment layout ---
// Fragment (kc,nt): lane holds N=nt*16+(lane&15), K = kc*32 + (lane>>4)*8 + (j&7) + (j>>3)*16
// stored as dst[((frag*32+lane)*16)+j]  -> loader does one 32B load per lane.
__global__ void pack_wmma_b_kernel(const float* __restrict__ W,
                                   _Float16* __restrict__ dst,
                                   int Kdim, int Ndim, int NT, int total) {
  int tid = blockIdx.x * blockDim.x + threadIdx.x;
  if (tid >= total) return;
  int j    = tid & 15;
  int lane = (tid >> 4) & 31;
  int frag = tid >> 9;
  int kc = frag / NT;
  int nt = frag - kc * NT;
  int k  = kc * 32 + ((lane >> 4) << 3) + (j & 7) + ((j >> 3) << 4);
  int n  = nt * 16 + (lane & 15);
  float v = (k < Kdim && n < Ndim) ? W[(size_t)k * Ndim + n] : 0.0f;
  dst[tid] = (_Float16)v;
}

// -------- WMMA tile-group compute ------------------------------------------
template<int KC, int NTW>
__device__ __forceinline__ void wmma_tiles(const _Float16* src,
                                           const _Float16* wp,
                                           int ntBase, int NTtot, int lane,
                                           v8f (&acc)[NTW]) {
  #pragma unroll
  for (int j = 0; j < NTW; ++j) {
    v8f z = {0.f,0.f,0.f,0.f,0.f,0.f,0.f,0.f};
    acc[j] = z;
  }
  #pragma unroll
  for (int kc = 0; kc < KC; ++kc) {
    v16h a = *(const v16h*)(src + (kc << 9) + (lane << 4));   // 32B ds load
    #pragma unroll
    for (int j = 0; j < NTW; ++j) {
      const _Float16* bp = wp + ((size_t)((kc * NTtot + ntBase + j) * 32 + lane) << 4);
      v16h b = *(const v16h*)bp;                              // 32B L2-hot load
      acc[j] = __builtin_amdgcn_wmma_f32_16x16x32_f16(
          false, a, false, b, (short)0, acc[j], false, false);
    }
  }
}

// -------- bias + LayerNorm(256) + GELU + store to A-layout LDS -------------
// C/D layout: lane l, vgpr i -> row M = i + 8*(l>>4), col N = nt*16 + (l&15)
// gamma/beta/bias are preloaded in registers (loop-invariant per wave).
template<int NTW>
__device__ __forceinline__ void ln_gelu_store(v8f (&acc)[NTW],
                                              const float (&pb)[NTW],
                                              const float (&pg)[NTW],
                                              const float (&pbe)[NTW],
                                              _Float16* dst,
                                              int ntBase, int lane, int wave, int tid,
                                              float* sS, float* sQ,
                                              float* sM, float* sR) {
  const int nl = lane & 15;
  const int hf = lane >> 4;
  #pragma unroll
  for (int j = 0; j < NTW; ++j) {
    #pragma unroll
    for (int i = 0; i < 8; ++i) acc[j][i] += pb[j];
  }
  // per-wave partial row stats over this wave's NTW*16 columns
  float s[8], q[8];
  #pragma unroll
  for (int i = 0; i < 8; ++i) {
    float ss = 0.f, qq = 0.f;
    #pragma unroll
    for (int j = 0; j < NTW; ++j) { float v = acc[j][i]; ss += v; qq += v * v; }
    #pragma unroll
    for (int msk = 1; msk < 16; msk <<= 1) {
      ss += __shfl_xor(ss, msk, 32);
      qq += __shfl_xor(qq, msk, 32);
    }
    s[i] = ss; q[i] = qq;
  }
  if (nl == 0) {
    #pragma unroll
    for (int i = 0; i < 8; ++i) {
      int row = i + 8 * hf;
      sS[row * NWAVES + wave] = s[i];
      sQ[row * NWAVES + wave] = q[i];
    }
  }
  __syncthreads();
  // 16 designated threads finalize per-row mean / rstd
  if (tid < 16) {
    float ss = 0.f, qq = 0.f;
    #pragma unroll
    for (int w = 0; w < NWAVES; ++w) {
      ss += sS[tid * NWAVES + w];
      qq += sQ[tid * NWAVES + w];
    }
    float mu  = ss * (1.0f / 256.0f);
    float var = qq * (1.0f / 256.0f) - mu * mu;
    sM[tid] = mu;
    sR[tid] = rsqrtf(var + 1e-5f);
  }
  __syncthreads();
  float mean[8], rstd[8];
  #pragma unroll
  for (int i = 0; i < 8; ++i) {
    int row = i + 8 * hf;
    mean[i] = sM[row];
    rstd[i] = sR[row];
  }
  #pragma unroll
  for (int j = 0; j < NTW; ++j) {
    int n = (ntBase + j) * 16 + nl;
    float gv = pg[j], bev = pbe[j];
    #pragma unroll
    for (int i = 0; i < 8; ++i) {
      float v = (acc[j][i] - mean[i]) * rstd[i] * gv + bev;
      v = gelu_exact(v);
      dst[aaddr(i + 8 * hf, n)] = (_Float16)v;
    }
  }
  __syncthreads();  // dst complete before next layer reads it
}

// -------- 100-step flow integrator: one 16-row batch tile per block --------
__global__ __launch_bounds__(256, 1)
void flow_steps_kernel(const float* __restrict__ ctx,
                       const _Float16* __restrict__ w0p, const _Float16* __restrict__ w1p,
                       const _Float16* __restrict__ w2p, const _Float16* __restrict__ w3p,
                       const float* __restrict__ b0, const float* __restrict__ g0, const float* __restrict__ be0,
                       const float* __restrict__ b1, const float* __restrict__ g1, const float* __restrict__ be1,
                       const float* __restrict__ b2, const float* __restrict__ g2, const float* __restrict__ be2,
                       const float* __restrict__ b3,
                       const float* __restrict__ noise,
                       float* __restrict__ out) {
  __shared__ __align__(32) _Float16 sIn[12 * 512];  // concat [x|ctx|t|pad], A-layout
  __shared__ __align__(32) _Float16 sA[8 * 512];    // hidden ping
  __shared__ __align__(32) _Float16 sB[8 * 512];    // hidden pong
  __shared__ float sx[16 * FLATD];                  // fp32 state x
  __shared__ float sS[16 * NWAVES];                 // LN cross-wave sums
  __shared__ float sQ[16 * NWAVES];                 // LN cross-wave sq-sums
  __shared__ float sM[16];                          // LN per-row mean
  __shared__ float sR[16];                          // LN per-row rstd

  const int tid  = threadIdx.x;
  const int lane = tid & 31;
  const int wave = tid >> 5;
  const int nl   = lane & 15;
  const int hf   = lane >> 4;
  const int rowBase = blockIdx.x * 16;

  // preload loop-invariant LN params / biases for this wave's tiles
  float b0r[2], g0r[2], be0r[2];
  float b1r[2], g1r[2], be1r[2];
  float b2r[2], g2r[2], be2r[2];
  #pragma unroll
  for (int j = 0; j < 2; ++j) {
    int n = (wave * 2 + j) * 16 + nl;
    b0r[j] = b0[n]; g0r[j] = g0[n]; be0r[j] = be0[n];
    b1r[j] = b1[n]; g1r[j] = g1[n]; be1r[j] = be1[n];
    b2r[j] = b2[n]; g2r[j] = g2[n]; be2r[j] = be2[n];
  }
  const float b3r = (wave < 7) ? b3[wave * 16 + nl] : 0.f;

  // zero concat buffer (covers K padding 369..383)
  for (int i = tid; i < 12 * 512; i += 256) sIn[i] = (_Float16)0.f;
  __syncthreads();
  // ctx -> A-layout (features 112..367), written once
  for (int i = tid; i < 16 * AHD; i += 256) {
    int m = i >> 8, j = i & 255;
    sIn[aaddr(m, 112 + j)] = (_Float16)ctx[(size_t)(rowBase + m) * AHD + j];
  }
  // x0 = noise * sigma, fp32 in LDS
  for (int i = tid; i < 16 * FLATD; i += 256)
    sx[i] = noise[(size_t)rowBase * FLATD + i] * 0.1f;
  __syncthreads();

  const float dt = 1.0f / (float)STEPS;
  #pragma unroll 1
  for (int s = 0; s < STEPS; ++s) {
    asm volatile("" ::: "memory");  // keep weight-fragment loads inside the loop
    float t = (float)(STEPS - s) * dt;

    // refresh x and t in the concat buffer
    for (int i = tid; i < 16 * FLATD; i += 256) {
      int m = i / FLATD, k = i - m * FLATD;
      sIn[aaddr(m, k)] = (_Float16)sx[i];
    }
    if (tid < 16) sIn[aaddr(tid, 368)] = (_Float16)t;
    __syncthreads();

    v8f acc[2];
    // layer 0: K=384(padded), 16 N-tiles split 2/wave
    wmma_tiles<12, 2>(sIn, w0p, wave * 2, 16, lane, acc);
    ln_gelu_store<2>(acc, b0r, g0r, be0r, sA, wave * 2, lane, wave, tid, sS, sQ, sM, sR);
    // layer 1
    wmma_tiles<8, 2>(sA, w1p, wave * 2, 16, lane, acc);
    ln_gelu_store<2>(acc, b1r, g1r, be1r, sB, wave * 2, lane, wave, tid, sS, sQ, sM, sR);
    // layer 2
    wmma_tiles<8, 2>(sB, w2p, wave * 2, 16, lane, acc);
    ln_gelu_store<2>(acc, b2r, g2r, be2r, sA, wave * 2, lane, wave, tid, sS, sQ, sM, sR);

    // layer 3: 112 cols padded to 8 N-tiles, 1/wave (wave 7 idle: zero pad)
    if (wave < 7) {
      v8f p[1];
      wmma_tiles<8, 1>(sA, w3p, wave, 8, lane, p);
      // Euler update: x <- x - (dt/max(t,.05)) * (pred - x)
      float inv = dt / fmaxf(t, 0.05f);
      #pragma unroll
      for (int i = 0; i < 8; ++i) {
        int m = i + 8 * hf;
        int col = wave * 16 + nl;
        float xv = sx[m * FLATD + col];
        float pr = p[0][i] + b3r;
        sx[m * FLATD + col] = xv - inv * (pr - xv);
      }
    }
    __syncthreads();
  }

  for (int i = tid; i < 16 * FLATD; i += 256)
    out[(size_t)rowBase * FLATD + i] = sx[i];
}

// -------- context MLPs (~1% of FLOPs): plain fp32, one row per block -------
__device__ __forceinline__ void ln_stats_256(float v, float* red, float& mu, float& var) {
  float s = v, q = v * v;
  #pragma unroll
  for (int m = 16; m > 0; m >>= 1) {
    s += __shfl_xor(s, m, 32);
    q += __shfl_xor(q, m, 32);
  }
  int w = threadIdx.x >> 5;
  __syncthreads();
  if ((threadIdx.x & 31) == 0) { red[w] = s; red[8 + w] = q; }
  __syncthreads();
  s = 0.f; q = 0.f;
  #pragma unroll
  for (int i = 0; i < 8; ++i) { s += red[i]; q += red[8 + i]; }
  mu  = s * (1.0f / 256.0f);
  var = q * (1.0f / 256.0f) - mu * mu;
}

__global__ __launch_bounds__(256, 2)
void ctx_kernel(const float* __restrict__ hs, const float* __restrict__ rsn,
                const float* __restrict__ fw1, const float* __restrict__ fb1,
                const float* __restrict__ fg1, const float* __restrict__ fbe1,
                const float* __restrict__ fw2, const float* __restrict__ fb2,
                const float* __restrict__ rw1, const float* __restrict__ rb1,
                const float* __restrict__ rg1, const float* __restrict__ rbe1,
                const float* __restrict__ rw2, const float* __restrict__ rb2,
                float* __restrict__ ctx) {
  int b = blockIdx.x, n = threadIdx.x;
  __shared__ float shin[768];
  __shared__ float sh1[256];
  __shared__ float shr[8];
  __shared__ float red[16];
  const float* hrow = hs + ((size_t)b * 64 + 63) * 768;  // last token
  for (int k = n; k < 768; k += 256) shin[k] = hrow[k];
  if (n < 8) shr[n] = rsn[(size_t)b * 8 + n];
  __syncthreads();

  float a = fb1[n];
  for (int k = 0; k < 768; ++k) a += shin[k] * fw1[k * 256 + n];
  float mu, var;
  ln_stats_256(a, red, mu, var);
  float h = gelu_exact((a - mu) * rsqrtf(var + 1e-5f) * fg1[n] + fbe1[n]);
  __syncthreads();
  sh1[n] = h;
  __syncthreads();
  float c = fb2[n];
  for (int k = 0; k < 256; ++k) c += sh1[k] * fw2[k * 256 + n];

  float a2 = rb1[n];
  #pragma unroll
  for (int k = 0; k < 8; ++k) a2 += shr[k] * rw1[k * 256 + n];
  ln_stats_256(a2, red, mu, var);
  float h2 = gelu_exact((a2 - mu) * rsqrtf(var + 1e-5f) * rg1[n] + rbe1[n]);
  __syncthreads();
  sh1[n] = h2;
  __syncthreads();
  float r = rb2[n];
  for (int k = 0; k < 256; ++k) r += sh1[k] * rw2[k * 256 + n];

  ctx[(size_t)b * 256 + n] = c + r;
}

// ---------------------------------------------------------------------------
extern "C" void kernel_launch(void* const* d_in, const int* in_sizes, int n_in,
                              void* d_out, int out_size, void* d_ws, size_t ws_size,
                              hipStream_t stream) {
  (void)in_sizes; (void)n_in; (void)out_size; (void)ws_size;
  const float* hs    = (const float*)d_in[0];
  const float* rsn   = (const float*)d_in[1];
  const float* noise = (const float*)d_in[2];
  const float* fw1   = (const float*)d_in[3];
  const float* fb1   = (const float*)d_in[4];
  const float* fg1   = (const float*)d_in[5];
  const float* fbe1  = (const float*)d_in[6];
  const float* fw2   = (const float*)d_in[7];
  const float* fb2   = (const float*)d_in[8];
  const float* rw1   = (const float*)d_in[9];
  const float* rb1   = (const float*)d_in[10];
  const float* rg1   = (const float*)d_in[11];
  const float* rbe1  = (const float*)d_in[12];
  const float* rw2   = (const float*)d_in[13];
  const float* rb2   = (const float*)d_in[14];
  const float* vw0   = (const float*)d_in[15];
  const float* vb0   = (const float*)d_in[16];
  const float* vg0   = (const float*)d_in[17];
  const float* vbe0  = (const float*)d_in[18];
  const float* vw1   = (const float*)d_in[19];
  const float* vb1   = (const float*)d_in[20];
  const float* vg1   = (const float*)d_in[21];
  const float* vbe1  = (const float*)d_in[22];
  const float* vw2   = (const float*)d_in[23];
  const float* vb2   = (const float*)d_in[24];
  const float* vg2   = (const float*)d_in[25];
  const float* vbe2  = (const float*)d_in[26];
  const float* vw3   = (const float*)d_in[27];
  const float* vb3   = (const float*)d_in[28];

  unsigned char* ws = (unsigned char*)d_ws;
  // workspace layout
  float*     ctx = (float*)(ws + 0);                       // 2048*256*4 = 2097152
  _Float16*  w0p = (_Float16*)(ws + 2097152);              // 98304 halfs
  _Float16*  w1p = (_Float16*)(ws + 2097152 + 196608);     // 65536 halfs
  _Float16*  w2p = (_Float16*)(ws + 2097152 + 327680);     // 65536 halfs
  _Float16*  w3p = (_Float16*)(ws + 2097152 + 458752);     // 32768 halfs (N padded to 128)

  // pack weights into WMMA B-fragment layout (one-time per launch)
  pack_wmma_b_kernel<<<384, 256, 0, stream>>>(vw0, w0p, 369, 256, 16, 98304);
  pack_wmma_b_kernel<<<256, 256, 0, stream>>>(vw1, w1p, 256, 256, 16, 65536);
  pack_wmma_b_kernel<<<256, 256, 0, stream>>>(vw2, w2p, 256, 256, 16, 65536);
  pack_wmma_b_kernel<<<128, 256, 0, stream>>>(vw3, w3p, 256, 112,  8, 32768);

  // context MLPs
  ctx_kernel<<<2048, 256, 0, stream>>>(hs, rsn, fw1, fb1, fg1, fbe1, fw2, fb2,
                                       rw1, rb1, rg1, rbe1, rw2, rb2, ctx);

  // 100-step flow integration (WMMA): 128 blocks x 8 waves, 16 rows/block
  flow_steps_kernel<<<128, 256, 0, stream>>>(
      ctx, w0p, w1p, w2p, w3p,
      vb0, vg0, vbe0, vb1, vg1, vbe1, vb2, vg2, vbe2, vb3,
      noise, (float*)d_out);
}
